// Receiver_82489141887584
// MI455X (gfx1250) — matrix operand order repaired and checked by reading, not compiled, not run on hardware
//
#include <hip/hip_runtime.h>
#include <hip/hip_bf16.h>
#include <math.h>

// Problem constants (match reference)
#define BB   4096
#define NN   32
#define DIN  256
#define EE   128
#define HH   512
#define VV   32
#define IN0  384   // 3*E

typedef __attribute__((ext_vector_type(16))) __bf16 v16bf;
typedef __attribute__((ext_vector_type(8)))  float  v8f;
typedef __attribute__((ext_vector_type(4)))  int    v4i;

__device__ __forceinline__ unsigned short f2bfu(float f) {
  union { __bf16 h; unsigned short u; } cv; cv.h = (__bf16)f; return cv.u;
}
__device__ __forceinline__ __bf16 u2bf(unsigned short u) {
  union { unsigned short u; __bf16 h; } cv; cv.u = u; return cv.h;
}

// --------------------------- async global->LDS ------------------------------
#if defined(__has_builtin)
#if __has_builtin(__builtin_amdgcn_global_load_async_to_lds_b128)
#define HAVE_ASYNC_LDS 1
#endif
#endif

#ifdef HAVE_ASYNC_LDS
// Pointer-to-v4i in the global (1) and LDS (3) address spaces, matching the
// builtin's parameter types (diagnostic showed 'int __vector(4) *').
typedef __attribute__((address_space(1))) v4i* gv4i_p;
typedef __attribute__((address_space(3))) v4i* lv4i_p;

__device__ __forceinline__ void cp16_async(const float* g, float* l) {
  __builtin_amdgcn_global_load_async_to_lds_b128(
      (gv4i_p)(uintptr_t)g, (lv4i_p)(uintptr_t)l, 0, 0);
}
__device__ __forceinline__ void wait_async0() {
#if __has_builtin(__builtin_amdgcn_s_wait_asynccnt)
  __builtin_amdgcn_s_wait_asynccnt(0);
#else
  asm volatile("s_wait_asynccnt 0" ::: "memory");
#endif
}
#endif

// ---------------------------------------------------------------------------
// Attention: keys = cand[b] @ Wk + bk ; softmax(sender . keys) ; context ;
// x0[b] = concat(context, sender, prev_msg).  One block (128 thr / 4 waves)
// per batch row.  WMMA bf16 for the 32x128 = (32x256)x(256x128) GEMM.
// ---------------------------------------------------------------------------
__global__ void attention_kernel(const float* __restrict__ cand,
                                 const float* __restrict__ Wk,
                                 const float* __restrict__ bk,
                                 const float* __restrict__ sender,
                                 const float* __restrict__ prevmsg,
                                 float* __restrict__ x0) {
  __shared__ unsigned short candB[NN * DIN];   // 32x256 bf16 = 16 KB
  __shared__ float keys[NN * EE];              // 32x128 f32  = 16 KB
  __shared__ float red[128];
  __shared__ float wgt[NN];

  const int b   = blockIdx.x;
  const int tid = threadIdx.x;

  for (int i = tid; i < NN * DIN; i += 128)
    candB[i] = f2bfu(cand[(long)b * NN * DIN + i]);
  __syncthreads();

  const int lane = tid & 31, wave = tid >> 5;
  const int half = lane >> 4, ln16 = lane & 15;

  for (int t = wave; t < 16; t += 4) {
    const int mt = t >> 3, nt = t & 7;
    const int col = nt * 16 + ln16;
    v8f acc = {};
    for (int kt = 0; kt < DIN / 32; ++kt) {
      const int kb = kt * 32;
      v16bf a, bm;
      const unsigned short* pa = &candB[(mt * 16 + ln16) * DIN + kb + 8 * half];
#pragma unroll
      for (int i = 0; i < 8; ++i) a[i] = u2bf(pa[i]);
#pragma unroll
      for (int i = 0; i < 8; ++i) a[8 + i] = u2bf(pa[16 + i]);
      const float* pw = &Wk[(kb + 8 * half) * EE + col];
#pragma unroll
      for (int i = 0; i < 8; ++i) { bm[i] = (__bf16)pw[i * EE]; bm[8 + i] = (__bf16)pw[(16 + i) * EE]; }
      acc = __builtin_amdgcn_wmma_f32_16x16x32_bf16(false, a, false, bm,
                                                    (short)0, acc, false, false);
    }
    const float bv = bk[col];
#pragma unroll
    for (int r = 0; r < 8; ++r)
      keys[(mt * 16 + r + 8 * half) * EE + col] = acc[r] + bv;
  }
  __syncthreads();

  {
    const int n = tid >> 2, q = tid & 3;
    const float* se = &sender[(long)b * EE];
    float p = 0.f;
    for (int e = q * 32; e < q * 32 + 32; ++e) p += se[e] * keys[n * EE + e];
    red[tid] = p;
  }
  __syncthreads();
  if (tid < NN) {
    float s = red[tid * 4] + red[tid * 4 + 1] + red[tid * 4 + 2] + red[tid * 4 + 3];
    float m = s;
    for (int o = 16; o > 0; o >>= 1) m = fmaxf(m, __shfl_xor(m, o, 32));
    float e = expf(s - m);
    float sum = e;
    for (int o = 16; o > 0; o >>= 1) sum += __shfl_xor(sum, o, 32);
    wgt[tid] = e / sum;
  }
  __syncthreads();

  {
    const int e = tid;
    float c = 0.f;
#pragma unroll 4
    for (int n = 0; n < NN; ++n) c += wgt[n] * keys[n * EE + e];
    float* xr = &x0[(long)b * IN0];
    xr[e]          = c;
    xr[EE + e]     = sender[(long)b * EE + e];
    xr[2 * EE + e] = prevmsg[(long)b * EE + e];
  }
}

// ---------------------------------------------------------------------------
// Tiled GEMM: C[B x N] = A[B x K] @ W[K x N] + bias, bf16 WMMA, f32 acc.
// Block: 256 thr (8 waves) owns a 32(M) x 128(N) macro tile.
//   - A tile staged in LDS row-major bf16 (lane frag = 2x ds_load_b128)
//   - W k-slab (32 x ncols) staged cooperatively in LDS, col-major, pad 40
//     (16B-aligned columns -> lane B frag = 2x ds_load_b128), double-buffered.
// Grid: (B/32, ceil(N/128)).  K multiple of 32, K <= 512; N multiple of 16.
// ---------------------------------------------------------------------------
#define WPAD 40
__global__ void gemm_tiled_kernel(const float* __restrict__ A,
                                  const float* __restrict__ W,
                                  const float* __restrict__ bias,
                                  float* __restrict__ C,
                                  int K, int N) {
  __shared__ unsigned short As[32 * 512];        // 32 KB max
  __shared__ unsigned short Ws[2][128 * WPAD];   // 2 x 10 KB

  const int m0 = blockIdx.x * 32;
  const int n0 = blockIdx.y * 128;
  const int ncols = (N - n0 < 128) ? (N - n0) : 128;  // 128 or 32 here
  const int tid = threadIdx.x;

  // Stage A tile (no integer division in the hot loop)
  for (int r = 0; r < 32; ++r)
    for (int k = tid; k < K; k += 256)
      As[r * K + k] = f2bfu(A[(long)(m0 + r) * K + k]);

  // W staging thread mapping (ncols divides 256 for all our shapes)
  const int c_s     = tid % ncols;
  const int k0_s    = tid / ncols;
  const int kstride = 256 / ncols;
  auto stageW = [&](int kt, int buf) {
    const long kb = (long)kt * 32;
    for (int k = k0_s; k < 32; k += kstride)
      Ws[buf][c_s * WPAD + k] = f2bfu(W[(kb + k) * N + n0 + c_s]);
  };
  stageW(0, 0);
  __syncthreads();

  const int lane = tid & 31, wave = tid >> 5;
  const int half = lane >> 4, ln16 = lane & 15;
  const int c = wave * 16 + ln16;    // local output column
  const int ksteps = K >> 5;
  const bool active = (c < ncols);   // wave-uniform (ncols multiple of 16)

  v8f acc0 = {}, acc1 = {};
  for (int kt = 0; kt < ksteps; ++kt) {
    const int cur = kt & 1;
    if (kt + 1 < ksteps) stageW(kt + 1, cur ^ 1);   // overlap with compute
    if (active) {
      v16bf bm, a0, a1v;
      const unsigned short* pb  = &Ws[cur][c * WPAD + 8 * half];
#pragma unroll
      for (int i = 0; i < 8; ++i) { bm[i] = u2bf(pb[i]); bm[8 + i] = u2bf(pb[16 + i]); }
      const unsigned short* pa0 = &As[ln16 * K + kt * 32 + 8 * half];
      const unsigned short* pa1 = pa0 + 16 * K;
#pragma unroll
      for (int i = 0; i < 8; ++i) { a0[i] = u2bf(pa0[i]); a0[8 + i] = u2bf(pa0[16 + i]); }
#pragma unroll
      for (int i = 0; i < 8; ++i) { a1v[i] = u2bf(pa1[i]); a1v[8 + i] = u2bf(pa1[16 + i]); }
      acc0 = __builtin_amdgcn_wmma_f32_16x16x32_bf16(false, a0,  false, bm,
                                                     (short)0, acc0, false, false);
      acc1 = __builtin_amdgcn_wmma_f32_16x16x32_bf16(false, a1v, false, bm,
                                                     (short)0, acc1, false, false);
    }
    __syncthreads();
  }

  if (active) {
    const float bv = bias[n0 + c];
#pragma unroll
    for (int r = 0; r < 8; ++r) {
      C[(long)(m0 + r + 8 * half) * N + n0 + c]      = acc0[r] + bv;
      C[(long)(m0 + 16 + r + 8 * half) * N + n0 + c] = acc1[r] + bv;
    }
  }
}

// ---------------------------------------------------------------------------
// LayerNorm-GRU combine: one block (256 thr) per batch row.  Rows staged in
// LDS (async global->LDS DMA when available) and reused for stats + gating.
// ---------------------------------------------------------------------------
__device__ __forceinline__ void row_stats(const float* __restrict__ row, int n,
                                          float* rs, float* rq,
                                          float& mean, float& rstd) {
  const int t = threadIdx.x;
  float s = 0.f, q = 0.f;
  for (int j = t; j < n; j += 256) { const float v = row[j]; s += v; q += v * v; }
  rs[t] = s; rq[t] = q;
  __syncthreads();
  for (int o = 128; o > 0; o >>= 1) {
    if (t < o) { rs[t] += rs[t + o]; rq[t] += rq[t + o]; }
    __syncthreads();
  }
  mean = rs[0] / n;
  const float var = rq[0] / n - mean * mean;
  rstd = rsqrtf(var + 1e-5f);
  __syncthreads();
}

__global__ void gru_combine_kernel(const float* __restrict__ s1,
                                   const float* __restrict__ s2,
                                   const float* __restrict__ a1,
                                   const float* __restrict__ a2,
                                   const float* __restrict__ hprev,
                                   float* __restrict__ hout) {
  __shared__ float rb[3072];    // [s1:1024][s2:1024][a1:512][a2:512] = 12 KB
  __shared__ float rs[256], rq[256];
  const long b = blockIdx.x;
  const int  t = threadIdx.x;
  const float* S1g = s1 + b * (2 * HH);
  const float* S2g = s2 + b * (2 * HH);
  const float* A1g = a1 + b * HH;
  const float* A2g = a2 + b * HH;

#ifdef HAVE_ASYNC_LDS
  cp16_async(S1g + t * 4, &rb[t * 4]);
  cp16_async(S2g + t * 4, &rb[1024 + t * 4]);
  if (t < 128) {
    cp16_async(A1g + t * 4, &rb[2048 + t * 4]);
    cp16_async(A2g + t * 4, &rb[2560 + t * 4]);
  }
  wait_async0();
#else
  ((float4*)rb)[t]       = ((const float4*)S1g)[t];
  ((float4*)rb)[256 + t] = ((const float4*)S2g)[t];
  if (t < 128) {
    ((float4*)rb)[512 + t] = ((const float4*)A1g)[t];
    ((float4*)rb)[640 + t] = ((const float4*)A2g)[t];
  }
#endif
  __syncthreads();

  const float* S1 = rb;
  const float* S2 = rb + 1024;
  const float* A1 = rb + 2048;
  const float* A2 = rb + 2560;

  float m1, r1, m2, r2, m3, r3, m4, r4;
  row_stats(S1, 2 * HH, rs, rq, m1, r1);
  row_stats(S2, 2 * HH, rs, rq, m2, r2);
  row_stats(A1, HH,     rs, rq, m3, r3);
  row_stats(A2, HH,     rs, rq, m4, r4);

  for (int j = t; j < HH; j += 256) {
    const float z  = 1.f / (1.f + expf(-((S1[j] - m1) * r1 + (S2[j] - m2) * r2)));
    const float rt = 1.f / (1.f + expf(-((S1[HH + j] - m1) * r1 + (S2[HH + j] - m2) * r2)));
    const float hh = tanhf((A1[j] - m3) * r3 + rt * ((A2[j] - m4) * r4));
    const float h  = hprev[b * HH + j];
    hout[b * HH + j] = (1.f - z) * h + z * hh;
  }
}

// ---------------------------------------------------------------------------
// Head: V=32 == wave32 -> one wave per row. log_softmax, entropy, argmax.
// ---------------------------------------------------------------------------
__global__ void head_kernel(const float* __restrict__ logits,
                            float* __restrict__ tok,
                            float* __restrict__ ent,
                            float* __restrict__ tlg) {
  const int wave = threadIdx.x >> 5, lane = threadIdx.x & 31;
  const long row = (long)blockIdx.x * 8 + wave;
  const float x = logits[row * VV + lane];

  float m = x;
  for (int o = 16; o > 0; o >>= 1) m = fmaxf(m, __shfl_xor(m, o, 32));
  const float e = expf(x - m);
  float s = e;
  for (int o = 16; o > 0; o >>= 1) s += __shfl_xor(s, o, 32);
  const float lp = x - m - logf(s);
  const float p  = expf(lp);
  float pe = -p * lp;
  for (int o = 16; o > 0; o >>= 1) pe += __shfl_xor(pe, o, 32);

  float bv = x; int bi = lane;
  for (int o = 16; o > 0; o >>= 1) {
    const float ov = __shfl_xor(bv, o, 32);
    const int   oi = __shfl_xor(bi, o, 32);
    if (ov > bv || (ov == bv && oi < bi)) { bv = ov; bi = oi; }
  }
  const float tl = __shfl(lp, bi, 32);
  if (lane == 0) { tok[row] = (float)bi; ent[row] = pe; tlg[row] = tl; }
}

// ---------------------------------------------------------------------------
extern "C" void kernel_launch(void* const* d_in, const int* in_sizes, int n_in,
                              void* d_out, int out_size, void* d_ws, size_t ws_size,
                              hipStream_t stream) {
  (void)in_sizes; (void)n_in; (void)out_size; (void)ws_size;

  const float* sender = (const float*)d_in[0];
  const float* prevh  = (const float*)d_in[1];  // (2,B,H)
  const float* cand   = (const float*)d_in[2];
  const float* pmsg   = (const float*)d_in[3];
  const float* Wk     = (const float*)d_in[4];
  const float* bk     = (const float*)d_in[5];
  const float* i2h_W0 = (const float*)d_in[6];
  const float* i2h_b0 = (const float*)d_in[7];
  const float* h2h_W0 = (const float*)d_in[8];
  const float* h2h_b0 = (const float*)d_in[9];
  const float* hW_W0  = (const float*)d_in[10];
  const float* hW_b0  = (const float*)d_in[11];
  const float* hU_W0  = (const float*)d_in[12];
  const float* hU_b0  = (const float*)d_in[13];
  const float* i2h_W1 = (const float*)d_in[14];
  const float* i2h_b1 = (const float*)d_in[15];
  const float* h2h_W1 = (const float*)d_in[16];
  const float* h2h_b1 = (const float*)d_in[17];
  const float* hW_W1  = (const float*)d_in[18];
  const float* hW_b1  = (const float*)d_in[19];
  const float* hU_W1  = (const float*)d_in[20];
  const float* hU_b1  = (const float*)d_in[21];
  const float* out_W  = (const float*)d_in[22];
  const float* out_b  = (const float*)d_in[23];

  // Output layout: tok(B), entropy(B), tok_logits(B), logits(B*V), hidden(2*B*H)
  float* out    = (float*)d_out;
  float* tok    = out;
  float* ent    = out + BB;
  float* tlg    = out + 2L * BB;
  float* logits = out + 3L * BB;
  float* h0     = out + 3L * BB + (long)BB * VV;
  float* h1     = h0 + (long)BB * HH;

  // Workspace layout
  float* ws = (float*)d_ws;
  float* x0 = ws;                                // B x 384
  float* s1 = x0 + (long)BB * IN0;               // B x 1024
  float* s2 = s1 + (long)BB * 2 * HH;            // B x 1024
  float* a1 = s2 + (long)BB * 2 * HH;            // B x 512
  float* a2 = a1 + (long)BB * HH;                // B x 512

  const float* hp0 = prevh;
  const float* hp1 = prevh + (long)BB * HH;

  const dim3 blk(256);
  const dim3 g1024(BB / 32, 8);   // N = 1024
  const dim3 g512 (BB / 32, 4);   // N = 512
  const dim3 g32  (BB / 32, 1);   // N = 32

  // 1) attention + concat
  attention_kernel<<<BB, 128, 0, stream>>>(cand, Wk, bk, sender, pmsg, x0);

  // 2) GRU layer 0
  gemm_tiled_kernel<<<g1024, blk, 0, stream>>>(x0,  i2h_W0, i2h_b0, s1, IN0, 2 * HH);
  gemm_tiled_kernel<<<g1024, blk, 0, stream>>>(hp0, h2h_W0, h2h_b0, s2, HH,  2 * HH);
  gemm_tiled_kernel<<<g512,  blk, 0, stream>>>(x0,  hW_W0,  hW_b0,  a1, IN0, HH);
  gemm_tiled_kernel<<<g512,  blk, 0, stream>>>(hp0, hU_W0,  hU_b0,  a2, HH,  HH);
  gru_combine_kernel<<<BB, 256, 0, stream>>>(s1, s2, a1, a2, hp0, h0);

  // 3) GRU layer 1
  gemm_tiled_kernel<<<g1024, blk, 0, stream>>>(h0,  i2h_W1, i2h_b1, s1, HH, 2 * HH);
  gemm_tiled_kernel<<<g1024, blk, 0, stream>>>(hp1, h2h_W1, h2h_b1, s2, HH, 2 * HH);
  gemm_tiled_kernel<<<g512,  blk, 0, stream>>>(h0,  hW_W1,  hW_b1,  a1, HH, HH);
  gemm_tiled_kernel<<<g512,  blk, 0, stream>>>(hp1, hU_W1,  hU_b1,  a2, HH, HH);
  gru_combine_kernel<<<BB, 256, 0, stream>>>(s1, s2, a1, a2, hp1, h1);

  // 4) head
  gemm_tiled_kernel<<<g32, blk, 0, stream>>>(h1, out_W, out_b, logits, HH, VV);
  head_kernel<<<BB / 8, 256, 0, stream>>>(logits, tok, ent, tlg);
}
